// EnoughViT_63282048139394
// MI455X (gfx1250) — compile-verified, matrix operand order checked by reading, not run on hardware
//
#include <hip/hip_runtime.h>
#include <hip/hip_bf16.h>
#include <math.h>

// ---------------------------------------------------------------------------
// ViT forward for MI455X (gfx1250): bf16 WMMA GEMM core + fused epilogues.
// Wave32, 256-thread blocks (8 wave32s), 64x64 tiles, K-tile 64 in LDS,
// software-pipelined global->reg->LDS staging with vector (b128/b64) traffic.
// ---------------------------------------------------------------------------

typedef __attribute__((ext_vector_type(16))) __bf16 v16bf;
typedef __attribute__((ext_vector_type(8)))  float  v8f;
typedef __attribute__((ext_vector_type(4)))  __bf16 bf16x4;
typedef __attribute__((ext_vector_type(4)))  float  f32x4;

#define BM 64
#define BN 64
#define BKT 64     // K extent staged in LDS per iteration (2 WMMA k-steps)
#define ASTR 68    // LDS row stride in bf16 elems: 136B rows keep b64 stores aligned

// Generic batched GEMM: C[z] = act(alpha * A[z] @ B[z] + bias) + Res[z]
//  A: [M,K] row-major (lda, batch stride sA)
//  B: !transB -> [K,N] row-major; transB -> [N,K] row-major (ldb, stride sB)
//  Requirements met by all call sites: lda/ldb multiples of 4, 16B-aligned bases.
__global__ __launch_bounds__(256) void vit_wmma_gemm(
    const float* __restrict__ A, int lda, long sA,
    const float* __restrict__ B, int ldb, long sB, int transB,
    float* __restrict__ C, int ldc, long sC,
    const float* __restrict__ bias,
    const float* __restrict__ Res, int ldr, long sR,
    int M, int N, int K, float alpha, int act)
{
    __shared__ __bf16 As[BM * ASTR];          // A tile, row-major [m][k]
    __shared__ __bf16 Bs[BN * ASTR];          // B tile, TRANSPOSED [n][k]

    const int z = blockIdx.z;
    A += (long)z * sA;
    B += (long)z * sB;
    C += (long)z * sC;
    if (Res) Res += (long)z * sR;

    const int m0 = blockIdx.y * BM;
    const int n0 = blockIdx.x * BN;
    const int t    = threadIdx.x;
    const int lane = t & 31;
    const int wave = t >> 5;          // 0..7
    const int wm   = wave & 3;        // 16-row slab within 64
    const int wn   = wave >> 2;       // 0/1 -> 32-col slab within 64

    v8f acc[2];
    acc[0] = v8f{0.f,0.f,0.f,0.f,0.f,0.f,0.f,0.f};
    acc[1] = v8f{0.f,0.f,0.f,0.f,0.f,0.f,0.f,0.f};

    const int ml    = lane & 15;            // M (or N) index within 16-tile
    const int khalf = (lane >> 4) << 3;     // +8 K offset for upper half-wave

    // fragment build + 2 WMMAs per 32-wide k-subtile (ISA 7.12.2 16-bit layouts)
    auto mma_tile = [&]() {
        const __bf16* Ap  = &As[(16 * wm + ml) * ASTR];
        const __bf16* Bp0 = &Bs[(32 * wn + ml) * ASTR];
        const __bf16* Bp1 = &Bs[(32 * wn + 16 + ml) * ASTR];
        #pragma unroll
        for (int ks = 0; ks < BKT; ks += 32) {
            v16bf af, b0, b1;
            #pragma unroll
            for (int i = 0; i < 8; ++i) {
                int kb = ks + khalf + ((i < 4) ? (2 * i) : (16 + 2 * (i - 4)));
                af[2*i] = Ap[kb];  af[2*i+1] = Ap[kb+1];
                b0[2*i] = Bp0[kb]; b0[2*i+1] = Bp0[kb+1];
                b1[2*i] = Bp1[kb]; b1[2*i+1] = Bp1[kb+1];
            }
            acc[0] = __builtin_amdgcn_wmma_f32_16x16x32_bf16(
                         false, af, false, b0, (short)0, acc[0], false, false);
            acc[1] = __builtin_amdgcn_wmma_f32_16x16x32_bf16(
                         false, af, false, b1, (short)0, acc[1], false, false);
        }
    };

    const bool interior = (m0 + BM <= M) && (n0 + BN <= N) && ((K & (BKT - 1)) == 0);

    if (interior) {
        // -------- fast path: vector loads, reg staging, software pipeline ----
        const int ra = t >> 2;              // A row 0..63
        const int ca = (t & 3) << 4;        // A col base {0,16,32,48}
        const int nbt = t & 63;             // B(!trans): n lane
        const int kbt = (t >> 6) << 4;      // B(!trans): k base {0,16,32,48}
        const int nbr = t >> 2;             // B(trans): n row
        const int kbr = (t & 3) << 4;       // B(trans): k base

        f32x4 ar[4];
        float br[16];

        auto loadA = [&](int k0) {
            const float* p = A + (long)(m0 + ra) * lda + k0 + ca;
            ar[0] = *(const f32x4*)(p);
            ar[1] = *(const f32x4*)(p + 4);
            ar[2] = *(const f32x4*)(p + 8);
            ar[3] = *(const f32x4*)(p + 12);
        };
        auto loadB = [&](int k0) {
            if (!transB) {
                const float* p = B + (long)(k0 + kbt) * ldb + n0 + nbt;
                #pragma unroll
                for (int j = 0; j < 16; ++j) br[j] = p[(long)j * ldb];
            } else {
                const float* p = B + (long)(n0 + nbr) * ldb + k0 + kbr;
                #pragma unroll
                for (int j = 0; j < 4; ++j) {
                    f32x4 v = *(const f32x4*)(p + 4 * j);
                    br[4*j+0] = v.x; br[4*j+1] = v.y;
                    br[4*j+2] = v.z; br[4*j+3] = v.w;
                }
            }
        };
        auto stage = [&]() {
            #pragma unroll
            for (int j = 0; j < 4; ++j) {
                bf16x4 q = { (__bf16)ar[j].x, (__bf16)ar[j].y,
                             (__bf16)ar[j].z, (__bf16)ar[j].w };
                *(bf16x4*)&As[ra * ASTR + ca + 4 * j] = q;
            }
            const int nb = transB ? nbr : nbt;
            const int kb = transB ? kbr : kbt;
            #pragma unroll
            for (int j = 0; j < 4; ++j) {
                bf16x4 q = { (__bf16)br[4*j+0], (__bf16)br[4*j+1],
                             (__bf16)br[4*j+2], (__bf16)br[4*j+3] };
                *(bf16x4*)&Bs[nb * ASTR + kb + 4 * j] = q;
            }
        };

        loadA(0);
        loadB(0);
        for (int k0 = 0; k0 < K; k0 += BKT) {
            stage();
            __syncthreads();
            if (k0 + BKT < K) { loadA(k0 + BKT); loadB(k0 + BKT); }  // overlap with WMMA
            mma_tile();
            __syncthreads();
        }
    } else {
        // -------- edge path: branch-free clamped scalar staging --------------
        for (int k0 = 0; k0 < K; k0 += BKT) {
            for (int i = t; i < BM * BKT; i += 256) {
                int r = i >> 6, c = i & 63;
                int gm = m0 + r, gk = k0 + c;
                int sm = gm < M ? gm : M - 1;
                int sk = gk < K ? gk : K - 1;
                float v = A[(long)sm * lda + sk];
                if (gm >= M || gk >= K) v = 0.f;
                As[r * ASTR + c] = (__bf16)v;
            }
            for (int i = t; i < BN * BKT; i += 256) {
                int n = i >> 6, c = i & 63;
                int gk = k0 + c, gn = n0 + n;
                int sk = gk < K ? gk : K - 1;
                int sn = gn < N ? gn : N - 1;
                long idx = transB ? ((long)sn * ldb + sk) : ((long)sk * ldb + sn);
                float v = B[idx];
                if (gk >= K || gn >= N) v = 0.f;
                Bs[n * ASTR + c] = (__bf16)v;
            }
            __syncthreads();
            mma_tile();
            __syncthreads();
        }
    }

    // ---- fused epilogue: alpha, bias, GELU, residual ----
    const int mh = (lane >> 4) << 3;    // C layout: VGPR r -> M = r + 8*(lane>>4)
    #pragma unroll
    for (int f = 0; f < 2; ++f) {
        int gn = n0 + 32 * wn + 16 * f + ml;
        if (gn >= N) continue;
        float bb = bias ? bias[gn] : 0.f;
        #pragma unroll
        for (int r = 0; r < 8; ++r) {
            int gm = m0 + 16 * wm + r + mh;
            if (gm >= M) continue;
            float v = acc[f][r] * alpha + bb;
            if (act == 1) v = 0.5f * v * (1.f + erff(v * 0.70710678118654752f));
            if (Res) v += Res[(long)gm * ldr + gn];
            C[(long)gm * ldc + gn] = v;
        }
    }
}

// ---- LayerNorm over n elements, one block per row ----
__global__ __launch_bounds__(256) void vit_ln(
    const float* __restrict__ x, long instride,
    float* __restrict__ y, long outstride,
    const float* __restrict__ g, const float* __restrict__ b, int n)
{
    __shared__ float red[256];
    const float* xr = x + (long)blockIdx.x * instride;
    float* yr = y + (long)blockIdx.x * outstride;
    int t = threadIdx.x;
    float s = 0.f;
    for (int i = t; i < n; i += 256) s += xr[i];
    red[t] = s; __syncthreads();
    for (int o = 128; o > 0; o >>= 1) { if (t < o) red[t] += red[t + o]; __syncthreads(); }
    float mu = red[0] / n; __syncthreads();
    float v = 0.f;
    for (int i = t; i < n; i += 256) { float d = xr[i] - mu; v += d * d; }
    red[t] = v; __syncthreads();
    for (int o = 128; o > 0; o >>= 1) { if (t < o) red[t] += red[t + o]; __syncthreads(); }
    float rstd = rsqrtf(red[0] / n + 1e-5f); __syncthreads();
    for (int i = t; i < n; i += 256) yr[i] = (xr[i] - mu) * rstd * g[i] + b[i];
}

// ---- assemble xe[b,s,e] = (s<196 ? patch_embed : class_token) + pos ----
__global__ void vit_assemble(const float* __restrict__ tmp,
                             const float* __restrict__ ctok,
                             const float* __restrict__ pos,
                             float* __restrict__ xe, long total)
{
    long i = (long)blockIdx.x * 256 + threadIdx.x;
    if (i >= total) return;
    const long se = (long)197 * 768;
    long b = i / se, r = i % se;
    int s = (int)(r / 768), e = (int)(r % 768);
    float v = (s < 196) ? tmp[((long)b * 196 + s) * 768 + e] : ctok[e];
    xe[i] = v + pos[r];
}

// ---- first[s,e] = mean over batch of XWv[b,s,e] (stride S*E) ----
__global__ void vit_batchmean(const float* __restrict__ x, float* __restrict__ o,
                              long n, int nb, long stride)
{
    long i = (long)blockIdx.x * 256 + threadIdx.x;
    if (i >= n) return;
    float s = 0.f;
    for (int j = 0; j < nb; ++j) s += x[(long)j * stride + i];
    o[i] = s * (1.f / nb);
}

// ---- xe = hn + first(broadcast over batch) + second ----
__global__ void vit_combine(const float* __restrict__ hn, const float* __restrict__ fir,
                            const float* __restrict__ sec, float* __restrict__ xe,
                            long total, long se)
{
    long i = (long)blockIdx.x * 256 + threadIdx.x;
    if (i >= total) return;
    xe[i] = hn[i] + fir[i % se] + sec[i];
}

// ---- in-place log-softmax per row ----
__global__ __launch_bounds__(256) void vit_logsoftmax(float* __restrict__ x, int n)
{
    __shared__ float red[256];
    float* xr = x + (long)blockIdx.x * n;
    int t = threadIdx.x;
    float m = -1e30f;
    for (int i = t; i < n; i += 256) m = fmaxf(m, xr[i]);
    red[t] = m; __syncthreads();
    for (int o = 128; o > 0; o >>= 1) { if (t < o) red[t] = fmaxf(red[t], red[t + o]); __syncthreads(); }
    m = red[0]; __syncthreads();
    float s = 0.f;
    for (int i = t; i < n; i += 256) s += expf(xr[i] - m);
    red[t] = s; __syncthreads();
    for (int o = 128; o > 0; o >>= 1) { if (t < o) red[t] += red[t + o]; __syncthreads(); }
    float ls = logf(red[0]); __syncthreads();
    for (int i = t; i < n; i += 256) xr[i] = xr[i] - m - ls;
}

// ---------------------------------------------------------------------------
static inline void launch_gemm(hipStream_t st,
    const float* A, int lda, long sA,
    const float* B, int ldb, long sB, int tB,
    float* C, int ldc, long sC,
    const float* bias, const float* Res, int ldr, long sR,
    int M, int N, int K, float alpha, int act, int batch)
{
    dim3 g((N + BN - 1) / BN, (M + BM - 1) / BM, batch);
    vit_wmma_gemm<<<g, dim3(256), 0, st>>>(A, lda, sA, B, ldb, sB, tB,
                                           C, ldc, sC, bias, Res, ldr, sR,
                                           M, N, K, alpha, act);
}

extern "C" void kernel_launch(void* const* d_in, const int* in_sizes, int n_in,
                              void* d_out, int out_size, void* d_ws, size_t ws_size,
                              hipStream_t stream)
{
    (void)in_sizes; (void)n_in; (void)out_size; (void)ws_size;

    const int B = 64, S = 197, NP = 196, E = 768, H = 3072, L = 12, CLS = 1000;
    const long SE = (long)S * E;
    const long BSE = (long)B * SE;

    const float* x_in  = (const float*)d_in[0];
    const float* W_emb = (const float*)d_in[1];
    const float* b_emb = (const float*)d_in[2];
    const float* ctok  = (const float*)d_in[3];
    const float* pos   = (const float*)d_in[4];
    const float* ln1s  = (const float*)d_in[5];
    const float* ln1b  = (const float*)d_in[6];
    const float* ln2s  = (const float*)d_in[7];
    const float* ln2b  = (const float*)d_in[8];
    const float* Wv    = (const float*)d_in[9];
    const float* Wt    = (const float*)d_in[10];
    const float* W1    = (const float*)d_in[11];
    const float* b1    = (const float*)d_in[12];
    const float* W2    = (const float*)d_in[13];
    const float* b2    = (const float*)d_in[14];
    const float* lnfs  = (const float*)d_in[15];
    const float* lnfb  = (const float*)d_in[16];
    const float* Wc1   = (const float*)d_in[17];
    const float* bc1   = (const float*)d_in[18];
    const float* Wc2   = (const float*)d_in[19];
    const float* bc2   = (const float*)d_in[20];
    float* out = (float*)d_out;

    // workspace carve-up (f32); all offsets 16B-aligned
    float* w = (float*)d_ws;
    float* tmp = w;  w += (size_t)B * NP * E;   // patch embed out [12544,768]
    float* xe  = w;  w += (size_t)BSE;          // activations [B,S,E]
    float* hn  = w;  w += (size_t)BSE;          // LN output (reused for LN2)
    float* xwv = w;  w += (size_t)BSE;
    float* xwt = w;  w += (size_t)BSE;
    float* sec = w;  w += (size_t)BSE;
    float* sco = w;  w += (size_t)S * B * B;    // scores [S,64,64]
    float* fir = w;  w += (size_t)SE;           // batch mean [S,E]
    float* h1  = w;  w += (size_t)B * S * H;    // MLP hidden
    float* cls = w;  w += (size_t)B * E;
    float* g1  = w;  w += (size_t)B * H;

    const float alpha2 = (1.0f / sqrtf((float)E)) / (float)B;  // inv_sqrt_d / n

    // 1) patch embedding: flat reshape of x is exactly [12544,768]
    launch_gemm(stream, x_in, E, 0, W_emb, E, 0, 0, tmp, E, 0,
                b_emb, nullptr, 0, 0, B * NP, E, E, 1.f, 0, 1);
    vit_assemble<<<dim3((BSE + 255) / 256), dim3(256), 0, stream>>>(tmp, ctok, pos, xe, BSE);

    // 2) encoder layers
    for (int l = 0; l < L; ++l) {
        vit_ln<<<dim3(B * S), dim3(256), 0, stream>>>(xe, E, hn, E,
                                                      ln1s + (long)l * E, ln1b + (long)l * E, E);
        // XWv, XWtheta : [B*S,768] @ [768,768]
        launch_gemm(stream, hn, E, 0, Wv + (long)l * E * E, E, 0, 0,
                    xwv, E, 0, nullptr, nullptr, 0, 0, B * S, E, E, 1.f, 0, 1);
        launch_gemm(stream, hn, E, 0, Wt + (long)l * E * E, E, 0, 0,
                    xwt, E, 0, nullptr, nullptr, 0, 0, B * S, E, E, 1.f, 0, 1);
        // scores[s] = X_s @ (X_s Wt)^T  (batched over s, rows strided by S*E)
        launch_gemm(stream, hn, (int)SE, E, xwt, (int)SE, E, /*transB=*/1,
                    sco, B, (long)B * B, nullptr, nullptr, 0, 0, B, B, E, 1.f, 0, S);
        // first[s,e] = mean over batch of XWv
        vit_batchmean<<<dim3((int)((SE + 255) / 256)), dim3(256), 0, stream>>>(xwv, fir, SE, B, SE);
        // second[s] = alpha2 * scores[s] @ XWv_s  (written into [B,S,E] layout)
        launch_gemm(stream, sco, B, (long)B * B, xwv, (int)SE, E, 0,
                    sec, (int)SE, E, nullptr, nullptr, 0, 0, B, E, B, alpha2, 0, S);
        // xe = hn + first + second   (residual is around norm1 output per reference)
        vit_combine<<<dim3((int)((BSE + 255) / 256)), dim3(256), 0, stream>>>(hn, fir, sec, xe, BSE, SE);
        // MLP with fused GELU / residual
        vit_ln<<<dim3(B * S), dim3(256), 0, stream>>>(xe, E, hn, E,
                                                      ln2s + (long)l * E, ln2b + (long)l * E, E);
        launch_gemm(stream, hn, E, 0, W1 + (long)l * E * H, H, 0, 0,
                    h1, H, 0, b1 + (long)l * H, nullptr, 0, 0, B * S, H, E, 1.f, /*gelu*/1, 1);
        launch_gemm(stream, h1, H, 0, W2 + (long)l * H * E, E, 0, 0,
                    xe, E, 0, b2 + (long)l * E, /*Res=*/xe, E, 0, B * S, E, H, 1.f, 0, 1);
    }

    // 3) classifier head on class token (LAST sequence position)
    vit_ln<<<dim3(B), dim3(256), 0, stream>>>(xe + (long)NP * E, SE, cls, E, lnfs, lnfb, E);
    launch_gemm(stream, cls, E, 0, Wc1, H, 0, 0, g1, H, 0,
                bc1, nullptr, 0, 0, B, H, E, 1.f, /*gelu*/1, 1);
    launch_gemm(stream, g1, H, 0, Wc2, CLS, 0, 0, out, CLS, 0,
                bc2, nullptr, 0, 0, B, CLS, H, 1.f, 0, 1);
    vit_logsoftmax<<<dim3(B), dim3(256), 0, stream>>>(out, CLS);
}
